// RNN_44246753083896
// MI455X (gfx1250) — matrix operand order repaired and checked by reading, not compiled, not run on hardware
//
#include <hip/hip_runtime.h>
#include <math.h>

#define N        4096
#define TSTEPS   2048
#define LD_OUT   (TSTEPS + 1)     // 2049
#define KTILES   (N / 32)         // 128 k-tiles of 32
#define NBLOCKS  (N / 16)         // 256 row-blocks, 16 rows each
#define SLICE_BYTES (16 * N * 2)  // 131072 bytes of bf16 W per block

typedef __attribute__((ext_vector_type(16))) __bf16 v16bf;
typedef __attribute__((ext_vector_type(8)))  float  v8f;
typedef __attribute__((ext_vector_type(4)))  unsigned uint32x4;
typedef __attribute__((ext_vector_type(8)))  int      int32x8;
typedef __attribute__((ext_vector_type(4)))  int      int32x4;

__device__ __forceinline__ unsigned short f32_to_bf16(float f) {
  union { float f; unsigned u; } c; c.f = f;
  unsigned u = c.u;
  unsigned rnd = 0x7FFFu + ((u >> 16) & 1u);   // round-to-nearest-even
  return (unsigned short)((u + rnd) >> 16);
}

// ---------------------------------------------------------------------------
// One-time repack: W fp32 row-major -> bf16 tiles in WMMA A-fragment order.
// Tile (tr, kt) covers rows [tr*16, tr*16+16), K [kt*32, kt*32+32).
// Contiguous k order used here (B reads r the same way): lanes 0-15 hold row M=lane, K {0..15};
// lanes 16-31 hold row M=lane-16, K {16..31}. Each lane's 16 bf16 are
// stored contiguously (32 B) at Wp + tileIdx*512 + lane*16 shorts.
// ---------------------------------------------------------------------------
__global__ void repack_w(const float* __restrict__ W, unsigned short* __restrict__ Wp) {
  const int wave = (int)((blockIdx.x * blockDim.x + threadIdx.x) >> 5);
  const int lane = (int)(threadIdx.x & 31);
  const int tr   = wave >> 7;             // row-block 0..255
  const int kt   = wave & (KTILES - 1);   // k-tile    0..127
  const int row   = tr * 16 + (lane & 15);
  const int kbase = kt * 32 + ((lane >> 4) << 4);   // +0 (lanes 0-15) or +16 (16-31)
  const float* src = W + (size_t)row * N + kbase;
  unsigned short* dst = Wp + (size_t)wave * 512 + (size_t)lane * 16;
#pragma unroll
  for (int h = 0; h < 2; ++h)             // 8-element chunks: K +0 and +8
#pragma unroll
    for (int c = 0; c < 8; ++c)
      dst[h * 8 + c] = f32_to_bf16(src[h * 8 + c]);
}

// ---------------------------------------------------------------------------
// Per-launch init: R[:,0] = tanh(x0), seed bf16 r buffer 0, reset barrier.
// Must run every launch (d_ws state persists across graph replays).
// ---------------------------------------------------------------------------
__global__ void init_state(const float* __restrict__ x0, float* __restrict__ out,
                           unsigned short* __restrict__ rbf, unsigned* __restrict__ bar) {
  int i = (int)(blockIdx.x * blockDim.x + threadIdx.x);
  if (i < N) {
    float r = tanhf(x0[i]);
    out[(size_t)i * LD_OUT] = r;
    rbf[i] = f32_to_bf16(r);              // double-buffer slot 0 (read at t=0)
  }
  if (i == 0) { bar[0] = 0u; bar[1] = 0u; }   // counter, generation
}

// ---------------------------------------------------------------------------
// Persistent recurrence kernel: 256 blocks x 32 threads (1 wave32 each).
// Block b owns rows [b*16, b*16+16); its bf16 W slice (128 KB) lives in LDS
// for the whole kernel (filled by TDM tensor_load_to_lds, then overwritten by a
// plain copy). Per step: 128 x v_wmma_f32_16x16x32_bf16 across two
// independent accumulators, x/r update for the 16 owned rows, then a
// grid-wide sense-reversing barrier.
// ---------------------------------------------------------------------------
__global__ void __launch_bounds__(32)
rnn_persistent(const float* __restrict__ x0,
               const unsigned short* __restrict__ Wp,
               unsigned short* rbf,            // 2*N bf16, double-buffered
               float* __restrict__ out,        // (N, LD_OUT) row-major
               unsigned* bar) {
  __shared__ alignas(32) unsigned char ldsW[SLICE_BYTES];
  const int b    = (int)blockIdx.x;
  const int lane = (int)threadIdx.x;

  // ---- Stage this block's packed W slice (contiguous 128 KB) into LDS ----
  {
    // TDM: 2D tensor, 4-byte elements, one row of 32768 dwords.
    const unsigned long long gaddr =
        (unsigned long long)(const void*)(Wp + (size_t)b * (KTILES * 512));
    const unsigned lds_off = (unsigned)(unsigned long long)(void*)&ldsW[0];
    const unsigned ndw = SLICE_BYTES / 4;              // 32768 dwords

    uint32x4 g0;
    g0.x = 1u;                                         // count=1, is_restore=0
    g0.y = lds_off;                                    // bits 63:32  lds_addr
    g0.z = (unsigned)(gaddr & 0xFFFFFFFFull);          // bits 95:64  global_addr lo
    g0.w = (unsigned)((gaddr >> 32) & 0x01FFFFFFull)   // bits 120:96 global_addr hi
         | 0x80000000u;                                // bits 127:126 type=2
    int32x8 g1;
    g1[0] = (int)(2u << 16);                           // data_size=4B; mask/flags=0
    g1[1] = (int)((ndw & 0xFFFFu) << 16);              // tensor_dim0[15:0]
    g1[2] = (int)((1u << 16) | ((ndw >> 16) & 0xFFFFu)); // tensor_dim1=1 | dim0 hi
    g1[3] = (int)((ndw & 0xFFFFu) << 16);              // tile_dim0 | tensor_dim1 hi
    g1[4] = (int)1u;                                   // tile_dim1=1, tile_dim2=0
    g1[5] = (int)ndw;                                  // tensor_dim0_stride lo
    g1[6] = 0;                                         // stride hi / dim1_stride lo
    g1[7] = 0;
    int32x4 gz4 = {0, 0, 0, 0};
    int32x8 gz8 = {0, 0, 0, 0, 0, 0, 0, 0};
    // 6-arg form on this toolchain: (g0, g1, g2, g3, extra, cpol)
    __builtin_amdgcn_tensor_load_to_lds(g0, g1, gz4, gz4, gz8, 0);
    __builtin_amdgcn_s_wait_tensorcnt(0);

    // Belt-and-suspenders: plain copy of the same bytes (overwrites TDM
    // result; guarantees correctness independent of D# encoding details).
    const uint4* g = (const uint4*)(Wp + (size_t)b * (KTILES * 512));
    uint4* l = (uint4*)ldsW;
    for (int i = lane; i < SLICE_BYTES / 16; i += 32) l[i] = g[i];
  }
  __syncthreads();

  // Owner lanes: 0..7 handle rows m=0..7, lanes 16..23 handle rows m=8..15.
  const bool owner  = (lane < 8) || (lane >= 16 && lane < 24);
  const int  m      = (lane < 16) ? lane : (8 + (lane - 16));
  const int  row    = b * 16 + (m & 15);
  const int  accIdx = (lane < 16) ? lane : (lane - 16);  // 0..7 for owners
  float x = owner ? x0[row] : 0.0f;

  const v16bf* aw   = (const v16bf*)ldsW;   // 32 lanes * 32B per tile
  const int    bofs = lane >> 4;            // B-fragment half select

  for (int t = 0; t < TSTEPS; ++t) {
    const v16bf* rb = (const v16bf*)(rbf + (size_t)(t & 1) * N);
    v8f acc0 = {0.f, 0.f, 0.f, 0.f, 0.f, 0.f, 0.f, 0.f};
    v8f acc1 = {0.f, 0.f, 0.f, 0.f, 0.f, 0.f, 0.f, 0.f};

    // Two independent accumulators break the WMMA->WMMA RAW chain on C.
#pragma unroll 4
    for (int kt = 0; kt < KTILES; kt += 2) {
      v16bf a0 = aw[kt * 32 + lane];
      v16bf b0 = rb[kt * 2 + bofs];
      acc0 = __builtin_amdgcn_wmma_f32_16x16x32_bf16(
          false, a0, false, b0, (short)0, acc0, false, false);
      v16bf a1 = aw[(kt + 1) * 32 + lane];
      v16bf b1 = rb[(kt + 1) * 2 + bofs];
      acc1 = __builtin_amdgcn_wmma_f32_16x16x32_bf16(
          false, a1, false, b1, (short)0, acc1, false, false);
    }
    v8f acc = acc0 + acc1;

    if (owner) {
      // All D columns are equal; pick element accIdx via select chain
      // (avoid dynamic vector indexing -> scratch).
      float y = acc[0];
#pragma unroll
      for (int j = 1; j < 8; ++j) if (accIdx == j) y = acc[j];
      float xn = 0.8f * x + 0.3f * y;      // x + (1/tau)(-x + g*W*r), tau=5, g=1.5
      float r  = tanhf(xn);
      x = xn;
      out[(size_t)row * LD_OUT + (t + 1)] = r;
      rbf[(size_t)((t + 1) & 1) * N + row] = f32_to_bf16(r);
    }

    // Device-wide sense-reversing barrier between steps.
    __threadfence();
    if (lane == 0) {
      unsigned my  = __hip_atomic_load(&bar[1], __ATOMIC_ACQUIRE, __HIP_MEMORY_SCOPE_AGENT);
      unsigned got = __hip_atomic_fetch_add(&bar[0], 1u, __ATOMIC_ACQ_REL, __HIP_MEMORY_SCOPE_AGENT);
      if (got == gridDim.x - 1) {
        __hip_atomic_store(&bar[0], 0u, __ATOMIC_RELAXED, __HIP_MEMORY_SCOPE_AGENT);
        __hip_atomic_fetch_add(&bar[1], 1u, __ATOMIC_RELEASE, __HIP_MEMORY_SCOPE_AGENT);
      } else {
        while (__hip_atomic_load(&bar[1], __ATOMIC_ACQUIRE, __HIP_MEMORY_SCOPE_AGENT) == my)
          __builtin_amdgcn_s_sleep(2);
      }
    }
    __syncthreads();
  }
}

// ---------------------------------------------------------------------------
// d_in[0] = W (N*N f32), d_in[1] = x0 (N f32), d_in[2] = T (int, ignored: 2048)
// d_out   = R (N * (T+1) f32)
// d_ws    : [0, 32MB) packed bf16 W; [32MB, +16KB) r double buffer;
//           then 2 x u32 barrier state.  (needs ~33.6 MB)
// ---------------------------------------------------------------------------
extern "C" void kernel_launch(void* const* d_in, const int* in_sizes, int n_in,
                              void* d_out, int out_size, void* d_ws, size_t ws_size,
                              hipStream_t stream) {
  (void)in_sizes; (void)n_in; (void)out_size; (void)ws_size;
  const float* W  = (const float*)d_in[0];
  const float* x0 = (const float*)d_in[1];
  float* out = (float*)d_out;

  unsigned short* Wp  = (unsigned short*)d_ws;
  unsigned short* rbf = (unsigned short*)((char*)d_ws + (size_t)N * N * 2);
  unsigned*       bar = (unsigned*)((char*)d_ws + (size_t)N * N * 2 + (size_t)2 * N * 2);

  // 32768 tiles, one wave each, 8 waves per block -> 4096 blocks
  repack_w<<<(NBLOCKS * KTILES) / 8, 256, 0, stream>>>(W, Wp);
  init_state<<<(N + 255) / 256, 256, 0, stream>>>(x0, out, rbf, bar);
  rnn_persistent<<<NBLOCKS, 32, 0, stream>>>(x0, Wp, rbf, out, bar);
}
